// Att_Diffuse_model_6339371729750
// MI455X (gfx1250) — compile-verified, matrix operand order checked by reading, not run on hardware
//
#include <hip/hip_runtime.h>
#include <hip/hip_bf16.h>

// ---------------------------------------------------------------------------
// Types for CDNA5 WMMA (wave32): D(16x16 f32) = A(16x32 bf16) x B(32x16 bf16) + C
// ---------------------------------------------------------------------------
typedef __bf16 bf16;
typedef __attribute__((ext_vector_type(16))) __bf16 v16bf;
typedef __attribute__((ext_vector_type(8)))  float  v8f;
typedef int v4i_gcc __attribute__((vector_size(16))); // matches builtin param type

#define AS1 __attribute__((address_space(1)))
#define AS3 __attribute__((address_space(3)))

#if defined(__has_builtin)
#if __has_builtin(__builtin_amdgcn_global_load_async_to_lds_b128)
#define HAVE_ASYNC_LDS 1
#endif
#endif
#ifndef HAVE_ASYNC_LDS
#define HAVE_ASYNC_LDS 0
#endif

__device__ __forceinline__ void wait_asynccnt0() {
#if defined(__has_builtin)
#if __has_builtin(__builtin_amdgcn_s_wait_asynccnt)
  __builtin_amdgcn_s_wait_asynccnt(0);
  return;
#else
  asm volatile("s_wait_asynccnt 0" ::: "memory");
  return;
#endif
#else
  asm volatile("s_wait_asynccnt 0" ::: "memory");
#endif
}

__device__ __forceinline__ v8f wmma_bf16(v16bf a, v16bf b, v8f c) {
  return __builtin_amdgcn_wmma_f32_16x16x32_bf16(
      /*neg_a=*/false, a, /*neg_b=*/false, b,
      /*c_mod=*/(short)0, c, /*reuse_a=*/false, /*reuse_b=*/false);
}

// Model dims
#define BB   256
#define LL   200
#define DD   512
#define HH   8
#define DHH  64
#define TT   (BB * LL)        // 51200 tokens (multiple of 32)
#define LPAD 224              // L padded to multiple of 32 for attn*V k-loop
#define NEGF (-4294967295.0f) // -2^32 + 1
#define SCALE 0.04419417382415922f // 1/sqrt(512)

// ---------------------------------------------------------------------------
// block-wide sum reduction (256 threads)
// ---------------------------------------------------------------------------
__device__ __forceinline__ float block_reduce_sum(float v, float* red) {
  int tid = threadIdx.x;
  red[tid] = v;
  __syncthreads();
  for (int s = 128; s > 0; s >>= 1) {
    if (tid < s) red[tid] += red[tid + s];
    __syncthreads();
  }
  float r = red[0];
  __syncthreads();
  return r;
}

// ---------------------------------------------------------------------------
// 1) Weight convert: fp32 W[k][n] -> bf16 Wt[n][k]  (N-major for B fragments)
// ---------------------------------------------------------------------------
__global__ void wconvert_kernel(const float* __restrict__ W, bf16* __restrict__ Wt) {
  int idx = blockIdx.x * 256 + threadIdx.x;     // over 512*512
  int k = idx >> 9;
  int n = idx & 511;
  Wt[(size_t)n * DD + k] = (bf16)W[idx];
}

// ---------------------------------------------------------------------------
// 2) Fused embedding + LN0(eps 1e-12)*mask + LN1(eps 1e-5); emits masks
//    one block (256 thr) per token; each thread owns d=tid and d=tid+256
// ---------------------------------------------------------------------------
__global__ void embed_ln_kernel(const int* __restrict__ states,
                                const float* __restrict__ item_emb,
                                const float* __restrict__ pos_emb,
                                const float* __restrict__ ln0w, const float* __restrict__ ln0b,
                                const float* __restrict__ ln1w, const float* __restrict__ ln1b,
                                bf16* __restrict__ keys_bf,
                                bf16* __restrict__ queries_bf,
                                float* __restrict__ queries_f32,
                                float* __restrict__ key_mask,
                                float* __restrict__ query_mask,
                                float* __restrict__ maskv) {
  __shared__ float red[256];
  int token = blockIdx.x;
  int l = token % LL;
  int tid = threadIdx.x;
  size_t base = (size_t)token * DD;

  int st = states[token];
  const float* emb = item_emb + (size_t)st * DD;
  const float* pe  = pos_emb + (size_t)l * DD;
  float x0 = emb[tid] + pe[tid];
  float x1 = emb[tid + 256] + pe[tid + 256];

  keys_bf[base + tid]       = (bf16)x0;
  keys_bf[base + tid + 256] = (bf16)x1;

  float s  = block_reduce_sum(x0 + x1, red);
  float sq = block_reduce_sum(x0 * x0 + x1 * x1, red);
  float u   = s * (1.0f / DD);
  float var = fmaxf(sq * (1.0f / DD) - u * u, 0.0f);
  float inv = rsqrtf(var + 1e-12f);
  float m = (st > 0) ? 1.0f : 0.0f;
  float y0 = (ln0w[tid]       * (x0 - u) * inv + ln0b[tid])       * m;
  float y1 = (ln0w[tid + 256] * (x1 - u) * inv + ln0b[tid + 256]) * m;

  float s1  = block_reduce_sum(y0 + y1, red);
  float sq1 = block_reduce_sum(y0 * y0 + y1 * y1, red);
  float u1   = s1 * (1.0f / DD);
  float var1 = fmaxf(sq1 * (1.0f / DD) - u1 * u1, 0.0f);
  float inv1 = rsqrtf(var1 + 1e-5f);
  float q0 = ln1w[tid]       * (y0 - u1) * inv1 + ln1b[tid];
  float q1 = ln1w[tid + 256] * (y1 - u1) * inv1 + ln1b[tid + 256];

  queries_bf[base + tid]        = (bf16)q0;
  queries_bf[base + tid + 256]  = (bf16)q1;
  queries_f32[base + tid]       = q0;
  queries_f32[base + tid + 256] = q1;

  float qs = block_reduce_sum(q0 + q1, red);
  if (tid == 0) {
    key_mask[token]   = (s  != 0.0f) ? 1.0f : 0.0f; // sign(|sum keys|)
    query_mask[token] = (qs != 0.0f) ? 1.0f : 0.0f; // sign(|sum queries|)
    maskv[token] = m;
  }
}

// ---------------------------------------------------------------------------
// 3) WMMA GEMM: out[M,512] = X[M,512](bf16) @ W(bf16, stored Wt[n][k]) + bias
//    block = 8 waves; block owns one 32-row M tile (2 WMMA M-tiles);
//    wave w owns N-tiles 4w..4w+3 and shares each B fragment across both
//    M-tiles (8 WMMAs per k-step per wave). A slab staged to LDS via CDNA5
//    async global->LDS copy when the builtin is available.
// ---------------------------------------------------------------------------
__global__ void gemm_wmma_kernel(const bf16* __restrict__ X,
                                 const bf16* __restrict__ Wt,
                                 const float* __restrict__ bias,
                                 bf16* __restrict__ outB,
                                 float* __restrict__ outF,
                                 int relu) {
  __shared__ __attribute__((aligned(32))) bf16 as[32 * DD]; // 32 KB A slab
  int mt   = blockIdx.x;         // 32-row tile index
  int tid  = threadIdx.x;
  int lane = tid & 31;
  int wave = tid >> 5;
  int mrow  = lane & 15;
  int khalf = lane >> 4;

  // stage 32x512 bf16 (32 KB = 2048 x 16B) into LDS
  {
    const char* gsrc = (const char*)(X + (size_t)mt * 32 * DD);
#if HAVE_ASYNC_LDS
    for (int i = tid; i < 2048; i += 256) {
      __builtin_amdgcn_global_load_async_to_lds_b128(
          (AS1 v4i_gcc*)(gsrc + (size_t)i * 16),
          (AS3 v4i_gcc*)((AS3 char*)as + (size_t)i * 16),
          /*offset=*/0, /*cpol=*/0);
    }
    wait_asynccnt0();
#else
    const uint4* src = (const uint4*)gsrc;
    uint4* dst = (uint4*)as;
    for (int i = tid; i < 2048; i += 256) dst[i] = src[i];
#endif
  }
  __syncthreads();

  v8f acc[2][4] = {};
  for (int kk = 0; kk < 16; ++kk) {
    int kbase = kk * 32 + khalf * 16;
    v16bf a0 = *(const v16bf*)&as[(size_t)mrow * DD + kbase];
    v16bf a1 = *(const v16bf*)&as[(size_t)(mrow + 16) * DD + kbase];
#pragma unroll
    for (int t = 0; t < 4; ++t) {
      int n0 = (wave * 4 + t) * 16 + mrow;
      v16bf b = *(const v16bf*)(Wt + (size_t)n0 * DD + kbase);
      acc[0][t] = wmma_bf16(a0, b, acc[0][t]);
      acc[1][t] = wmma_bf16(a1, b, acc[1][t]);
    }
  }

#pragma unroll
  for (int mi = 0; mi < 2; ++mi) {
#pragma unroll
    for (int t = 0; t < 4; ++t) {
      int col = (wave * 4 + t) * 16 + mrow;
      float bsv = bias[col];
#pragma unroll
      for (int r = 0; r < 8; ++r) {
        int row = mt * 32 + mi * 16 + r + khalf * 8;
        float v = acc[mi][t][r] + bsv;
        if (relu) v = fmaxf(v, 0.0f);
        size_t idx = (size_t)row * DD + col;
        if (outB) outB[idx] = (bf16)v;
        if (outF) outF[idx] = v;
      }
    }
  }
}

// ---------------------------------------------------------------------------
// 4) V -> Vt[b,h,d,l] with L padded/zeroed to 224 (contiguous k for attn*V)
// ---------------------------------------------------------------------------
__global__ void vtrans_kernel(const bf16* __restrict__ V, bf16* __restrict__ Vt) {
  int idx = blockIdx.x * 256 + threadIdx.x;   // B*H*DH*LPAD
  if (idx >= BB * HH * DHH * LPAD) return;
  int l    = idx % LPAD;
  int rest = idx / LPAD;                      // (b*H+h)*DH + d
  int d  = rest & 63;
  int bh = rest >> 6;
  int h = bh & 7;
  int b = bh >> 3;
  float v = 0.0f;
  if (l < LL) v = (float)V[((size_t)(b * LL + l)) * DD + h * DHH + d];
  Vt[idx] = (bf16)v;
}

// ---------------------------------------------------------------------------
// 5) Attention: one block per (q-tile16, head, batch)
//    phase1: scores = Q Kt via WMMA -> LDS (f32)
//    phase2: mask + softmax + query-mask -> bf16 probs in LDS
//    phase3: out = P V via WMMA, + queries residual -> mh
// ---------------------------------------------------------------------------
__global__ void attn_kernel(const bf16* __restrict__ Qb,
                            const bf16* __restrict__ Kb,
                            const bf16* __restrict__ Vt,
                            const float* __restrict__ queries_f32,
                            const float* __restrict__ key_mask,
                            const float* __restrict__ query_mask,
                            float* __restrict__ mh) {
  __shared__ float sc[16 * 208];
  __shared__ __attribute__((aligned(32))) bf16 pb[16 * LPAD];
  __shared__ float red[256];

  int qt = blockIdx.x;   // 0..12
  int h  = blockIdx.y;   // 0..7
  int b  = blockIdx.z;   // 0..255
  int tid  = threadIdx.x;
  int lane = tid & 31;
  int wave = tid >> 5;
  int mrow  = lane & 15;
  int khalf = lane >> 4;

  // ---- phase 1: scores ----
  {
    int q  = qt * 16 + mrow;
    int qc = min(q, LL - 1);
    const bf16* qptr = Qb + ((size_t)(b * LL + qc)) * DD + h * DHH;
    for (int nt = wave; nt < 13; nt += 8) {
      int kcol = nt * 16 + mrow;
      int kc   = min(kcol, LL - 1);
      const bf16* kptr = Kb + ((size_t)(b * LL + kc)) * DD + h * DHH;
      v8f acc = {};
#pragma unroll
      for (int kk = 0; kk < 2; ++kk) {
        v16bf a = *(const v16bf*)(qptr + kk * 32 + khalf * 16);
        v16bf bmat = *(const v16bf*)(kptr + kk * 32 + khalf * 16);
        acc = wmma_bf16(a, bmat, acc);
      }
#pragma unroll
      for (int r = 0; r < 8; ++r) {
        int m = r + khalf * 8;                    // M row of tile
        sc[m * 208 + nt * 16 + mrow] = acc[r];    // N = mrow
      }
    }
  }
  __syncthreads();

  // ---- phase 2: masked softmax over n=0..199 per row ----
  {
    int row = tid >> 4;
    int sub = tid & 15;
    int qq  = qt * 16 + row;
    float qmv = (qq < LL) ? query_mask[b * LL + qq] : 0.0f;

    float pmx = -3.0e38f;
    for (int n = sub; n < 208; n += 16) {
      bool ok = (n < LL) && (n <= qq) && (key_mask[b * LL + n] != 0.0f);
      float v = ok ? sc[row * 208 + n] * SCALE : NEGF;
      sc[row * 208 + n] = v;
      pmx = fmaxf(pmx, v);
    }
    red[tid] = pmx;
    __syncthreads();
    if (sub == 0) {
      float m2 = red[tid];
      for (int j = 1; j < 16; ++j) m2 = fmaxf(m2, red[tid + j]);
      red[tid] = m2;
    }
    __syncthreads();
    float rowmax = red[row * 16];
    __syncthreads();

    float psum = 0.0f;
    for (int n = sub; n < LL; n += 16) {
      float e = __expf(sc[row * 208 + n] - rowmax);
      sc[row * 208 + n] = e;
      psum += e;
    }
    red[tid] = psum;
    __syncthreads();
    if (sub == 0) {
      float s2 = 0.0f;
      for (int j = 0; j < 16; ++j) s2 += red[tid + j];
      red[tid] = s2;
    }
    __syncthreads();
    float rinv = qmv / red[row * 16];

    for (int n = sub; n < LPAD; n += 16) {
      float p = (n < LL) ? sc[row * 208 + n] * rinv : 0.0f;
      pb[row * LPAD + n] = (bf16)p;
    }
  }
  __syncthreads();

  // ---- phase 3: out = P(16x224) @ Vt-slab(224x64), + residual ----
  if (wave < 4) {
    const bf16* vrow = Vt + (((size_t)(b * HH + h)) * DHH + wave * 16 + mrow) * LPAD;
    v8f acc = {};
#pragma unroll
    for (int kk = 0; kk < 7; ++kk) {
      v16bf a = *(const v16bf*)&pb[mrow * LPAD + kk * 32 + khalf * 16];
      v16bf bmat = *(const v16bf*)(vrow + kk * 32 + khalf * 16);
      acc = wmma_bf16(a, bmat, acc);
    }
#pragma unroll
    for (int r = 0; r < 8; ++r) {
      int m  = r + khalf * 8;
      int q2 = qt * 16 + m;
      if (q2 < LL) {
        size_t idx = ((size_t)(b * LL + q2)) * DD + h * DHH + wave * 16 + mrow;
        mh[idx] = acc[r] + queries_f32[idx];
      }
    }
  }
}

// ---------------------------------------------------------------------------
// 6) LN2 (eps 1e-5) -> x_f32 + x_bf16
// ---------------------------------------------------------------------------
__global__ void ln2_kernel(const float* __restrict__ mh,
                           const float* __restrict__ w, const float* __restrict__ bprm,
                           float* __restrict__ xf, bf16* __restrict__ xb) {
  __shared__ float red[256];
  int token = blockIdx.x;
  int tid = threadIdx.x;
  size_t base = (size_t)token * DD;
  float a0 = mh[base + tid];
  float a1 = mh[base + tid + 256];
  float s  = block_reduce_sum(a0 + a1, red);
  float sq = block_reduce_sum(a0 * a0 + a1 * a1, red);
  float u   = s * (1.0f / DD);
  float var = fmaxf(sq * (1.0f / DD) - u * u, 0.0f);
  float inv = rsqrtf(var + 1e-5f);
  float y0 = w[tid]       * (a0 - u) * inv + bprm[tid];
  float y1 = w[tid + 256] * (a1 - u) * inv + bprm[tid + 256];
  xf[base + tid]       = y0;
  xf[base + tid + 256] = y1;
  xb[base + tid]       = (bf16)y0;
  xb[base + tid + 256] = (bf16)y1;
}

// ---------------------------------------------------------------------------
// 7) Final: LNf(ff + x) -> *mask -> LN3 -> out
// ---------------------------------------------------------------------------
__global__ void ffn_final_kernel(const float* __restrict__ ff,
                                 const float* __restrict__ x,
                                 const float* __restrict__ maskv,
                                 const float* __restrict__ lnfw, const float* __restrict__ lnfb,
                                 const float* __restrict__ ln3w, const float* __restrict__ ln3b,
                                 float* __restrict__ out) {
  __shared__ float red[256];
  int token = blockIdx.x;
  int tid = threadIdx.x;
  size_t base = (size_t)token * DD;
  float a0 = ff[base + tid]       + x[base + tid];
  float a1 = ff[base + tid + 256] + x[base + tid + 256];
  float s  = block_reduce_sum(a0 + a1, red);
  float sq = block_reduce_sum(a0 * a0 + a1 * a1, red);
  float u   = s * (1.0f / DD);
  float var = fmaxf(sq * (1.0f / DD) - u * u, 0.0f);
  float inv = rsqrtf(var + 1e-5f);
  float m = maskv[token];
  float t0 = (lnfw[tid]       * (a0 - u) * inv + lnfb[tid])       * m;
  float t1 = (lnfw[tid + 256] * (a1 - u) * inv + lnfb[tid + 256]) * m;

  float s2  = block_reduce_sum(t0 + t1, red);
  float sq2 = block_reduce_sum(t0 * t0 + t1 * t1, red);
  float u2   = s2 * (1.0f / DD);
  float var2 = fmaxf(sq2 * (1.0f / DD) - u2 * u2, 0.0f);
  float inv2 = rsqrtf(var2 + 1e-5f);
  out[base + tid]       = ln3w[tid]       * (t0 - u2) * inv2 + ln3b[tid];
  out[base + tid + 256] = ln3w[tid + 256] * (t1 - u2) * inv2 + ln3b[tid + 256];
}

// ---------------------------------------------------------------------------
// Launch
// ---------------------------------------------------------------------------
extern "C" void kernel_launch(void* const* d_in, const int* in_sizes, int n_in,
                              void* d_out, int out_size, void* d_ws, size_t ws_size,
                              hipStream_t stream) {
  (void)in_sizes; (void)n_in; (void)out_size; (void)ws_size;

  const int*   states   = (const int*)d_in[0];
  // d_in[1] = len_states (unused by reference math)
  const float* item_emb = (const float*)d_in[2];
  const float* pos_emb  = (const float*)d_in[3];
  const float* ln0w = (const float*)d_in[4],  *ln0b = (const float*)d_in[5];
  const float* ln1w = (const float*)d_in[6],  *ln1b = (const float*)d_in[7];
  const float* ln2w = (const float*)d_in[8],  *ln2b = (const float*)d_in[9];
  const float* ln3w = (const float*)d_in[10], *ln3b = (const float*)d_in[11];
  const float* wq = (const float*)d_in[12], *bq = (const float*)d_in[13];
  const float* wk = (const float*)d_in[14], *bk = (const float*)d_in[15];
  const float* wv = (const float*)d_in[16], *bv = (const float*)d_in[17];
  const float* w1 = (const float*)d_in[18], *b1 = (const float*)d_in[19];
  const float* w2 = (const float*)d_in[20], *b2 = (const float*)d_in[21];
  const float* lnfw = (const float*)d_in[22], *lnfb = (const float*)d_in[23];
  float* out = (float*)d_out;
  char*  ws  = (char*)d_ws;

  const size_t f32TD  = (size_t)TT * DD * 4;   // 104,857,600
  const size_t bf16TD = (size_t)TT * DD * 2;   //  52,428,800
  const size_t vtSz   = (size_t)BB * HH * DHH * LPAD * 2; // 58,720,256
  const size_t wtSz   = (size_t)DD * DD * 2;   // 524,288

  size_t off = 0;
  size_t offA  = off; off += f32TD;   // queries_f32, later reused as x_f32
  size_t offBr = off; off += f32TD;   // mh, later reused as ff
  size_t offC  = off; off += bf16TD;  // keys_bf, later reused as x_bf
  size_t offDr = off; off += bf16TD;  // queries_bf, later reused as h1_bf
  size_t offE  = off; off += bf16TD;  // Q_bf
  size_t offF  = off; off += bf16TD;  // K_bf
  size_t offG  = off; off += bf16TD;  // V_bf
  size_t offH  = off; off += vtSz;    // Vt
  size_t offKM = off; off += (size_t)TT * 4;   // key_mask
  size_t offQM = off; off += (size_t)TT * 4;   // query_mask
  size_t offMK = off; off += (size_t)TT * 4;   // token mask
  size_t offW  = off; off += 5 * wtSz;         // 5 transposed bf16 weights

  float* queries_f32 = (float*)(ws + offA);
  float* x_f32       = (float*)(ws + offA);
  float* mh          = (float*)(ws + offBr);
  float* ffb         = (float*)(ws + offBr);
  bf16*  keys_bf     = (bf16*)(ws + offC);
  bf16*  x_bf        = (bf16*)(ws + offC);
  bf16*  queries_bf  = (bf16*)(ws + offDr);
  bf16*  h1_bf       = (bf16*)(ws + offDr);
  bf16*  Qb  = (bf16*)(ws + offE);
  bf16*  Kb  = (bf16*)(ws + offF);
  bf16*  Vb  = (bf16*)(ws + offG);
  bf16*  Vt  = (bf16*)(ws + offH);
  float* kmask = (float*)(ws + offKM);
  float* qmask = (float*)(ws + offQM);
  float* maskv = (float*)(ws + offMK);
  bf16* wqT = (bf16*)(ws + offW + 0 * wtSz);
  bf16* wkT = (bf16*)(ws + offW + 1 * wtSz);
  bf16* wvT = (bf16*)(ws + offW + 2 * wtSz);
  bf16* w1T = (bf16*)(ws + offW + 3 * wtSz);
  bf16* w2T = (bf16*)(ws + offW + 4 * wtSz);

  // 1) weight convert + transpose
  wconvert_kernel<<<1024, 256, 0, stream>>>(wq, wqT);
  wconvert_kernel<<<1024, 256, 0, stream>>>(wk, wkT);
  wconvert_kernel<<<1024, 256, 0, stream>>>(wv, wvT);
  wconvert_kernel<<<1024, 256, 0, stream>>>(w1, w1T);
  wconvert_kernel<<<1024, 256, 0, stream>>>(w2, w2T);

  // 2) embedding + LN0*mask + LN1
  embed_ln_kernel<<<TT, 256, 0, stream>>>(states, item_emb, pos_emb,
                                          ln0w, ln0b, ln1w, ln1b,
                                          keys_bf, queries_bf, queries_f32,
                                          kmask, qmask, maskv);

  // 3) Q/K/V projections (WMMA, 32-row blocks)
  gemm_wmma_kernel<<<TT / 32, 256, 0, stream>>>(queries_bf, wqT, bq, Qb, nullptr, 0);
  gemm_wmma_kernel<<<TT / 32, 256, 0, stream>>>(keys_bf,    wkT, bk, Kb, nullptr, 0);
  gemm_wmma_kernel<<<TT / 32, 256, 0, stream>>>(keys_bf,    wvT, bv, Vb, nullptr, 0);

  // 4) V transpose to [b,h,d,l] padded
  {
    int n = BB * HH * DHH * LPAD;
    vtrans_kernel<<<(n + 255) / 256, 256, 0, stream>>>(Vb, Vt);
  }

  // 5) attention (+ residual into mh)
  attn_kernel<<<dim3(13, HH, BB), 256, 0, stream>>>(Qb, Kb, Vt, queries_f32,
                                                    kmask, qmask, mh);

  // 6) LN2
  ln2_kernel<<<TT, 256, 0, stream>>>(mh, ln2w, ln2b, x_f32, x_bf);

  // 7) FFN GEMMs (WMMA): relu(x@w1+b1) @ w2 + b2
  gemm_wmma_kernel<<<TT / 32, 256, 0, stream>>>(x_bf,  w1T, b1, h1_bf, nullptr, 1);
  gemm_wmma_kernel<<<TT / 32, 256, 0, stream>>>(h1_bf, w2T, b2, nullptr, ffb, 0);

  // 8) LNf(ff+x) -> mask -> LN3 -> out
  ffn_final_kernel<<<TT, 256, 0, stream>>>(ffb, x_f32, maskv,
                                           lnfw, lnfb, ln3w, ln3b, out);
}
